// VmapCondensed_13597866459294
// MI455X (gfx1250) — compile-verified, matrix operand order checked by reading, not run on hardware
//
#include <hip/hip_runtime.h>
#include <stdint.h>

typedef __attribute__((ext_vector_type(16))) __bf16 v16bf;
typedef __attribute__((ext_vector_type(8)))  float  v8f;

#define B_    2048
#define IN_   4096
#define OUT_  4096
#define FAN_  64

// ================= WMMA GEMM path =================
// out[b,o] = sum_i Xbf[b,i] * ST[o,i] + bias[o]
#define TILE_M   128
#define TILE_N   128
#define TILE_K   32
#define KSTEPS   (IN_ / TILE_K)        // 128
#define ROWB     80                    // LDS bytes per tile row: 64B data + 16B pad
#define ATILE_B  (TILE_M * ROWB)       // 10240
#define BUF_B    (2 * ATILE_B)         // A + B tile per buffer
#define GEMM_LDS (2 * BUF_B)           // double buffered = 40960

// ---- pass 1: fp32 -> bf16 convert of the input activations ----
struct bf16x4 { __bf16 a, b, c, d; };
__global__ void cvt_bf16_kernel(const float* __restrict__ x,
                                __bf16* __restrict__ y, int n4) {
    int i = blockIdx.x * blockDim.x + threadIdx.x;
    if (i < n4) {
        float4 v = ((const float4*)x)[i];
        bf16x4 r = { (__bf16)v.x, (__bf16)v.y, (__bf16)v.z, (__bf16)v.w };
        ((bf16x4*)y)[i] = r;
    }
}

// ---- pass 2: scatter condensed weights into dense bf16 S_T[o][i] ----
// one thread owns one output row -> duplicate indices accumulate correctly, no atomics
__global__ void scatter_kernel(const long long* __restrict__ mask,
                               const float* __restrict__ w,
                               __bf16* __restrict__ ST) {
    int o = blockIdx.x * blockDim.x + threadIdx.x;
    if (o >= OUT_) return;
    __bf16* row = ST + (size_t)o * IN_;
    const long long* mi = mask + (size_t)o * FAN_;
    const float*     wi = w    + (size_t)o * FAN_;
    for (int f = 0; f < FAN_; ++f) {
        int i = (int)mi[f];
        row[i] = (__bf16)((float)row[i] + wi[f]);
    }
}

// ---- pass 3: dense bf16 GEMM via v_wmma_f32_16x16x32_bf16 ----
__global__ __launch_bounds__(256)
void wmma_gemm_kernel(const __bf16* __restrict__ X,   // [B_][IN_]
                      const __bf16* __restrict__ ST,  // [OUT_][IN_]
                      const float* __restrict__ bias,
                      float* __restrict__ out)
{
    extern __shared__ char smem[];
    const int tid = threadIdx.x;
    const int o0  = blockIdx.x * TILE_N;
    const int b0  = blockIdx.y * TILE_M;
    const int wv  = tid >> 5;            // wave 0..7
    const int l   = tid & 31;
    const int wm  = (wv >> 2) * 64;      // wave row origin in tile (2 waves along M)
    const int wn  = (wv & 3) * 32;       // wave col origin in tile (4 waves along N)

    v8f acc[4][2];
#pragma unroll
    for (int fm = 0; fm < 4; ++fm)
#pragma unroll
        for (int fn = 0; fn < 2; ++fn) acc[fm][fn] = (v8f)(0.f);

    // cooperative async stage of one k-slice (A: 128x32, B: 128x32) into buffer `buf`
    auto issue = [&](int buf, int kstep) {
        const int k0 = kstep * TILE_K;
        char* base = smem + buf * BUF_B;
#pragma unroll
        for (int it = 0; it < 2; ++it) {
            int i = tid + it * 256;          // 0..511 : 128 rows x 4 chunks of 16B
            int r = i >> 2, c = i & 3;
            const __bf16* gA = X  + (size_t)(b0 + r) * IN_ + k0 + c * 8;
            unsigned la = (unsigned)(uintptr_t)(base + r * ROWB + c * 16);
            asm volatile("global_load_async_to_lds_b128 %0, %1, off"
                         :: "v"(la), "v"(gA) : "memory");
            const __bf16* gB = ST + (size_t)(o0 + r) * IN_ + k0 + c * 8;
            unsigned lb = (unsigned)(uintptr_t)(base + ATILE_B + r * ROWB + c * 16);
            asm volatile("global_load_async_to_lds_b128 %0, %1, off"
                         :: "v"(lb), "v"(gB) : "memory");
        }
    };
    // fragment fetch: lane l -> row (l&15), K-half (l>>4); chunks at +0 and +32 bytes
    auto ldfrag = [&](const char* rowbase) -> v16bf {
        union { int4 q[2]; v16bf v; } u;
        u.q[0] = *(const int4*)(rowbase);
        u.q[1] = *(const int4*)(rowbase + 32);
        return u.v;
    };

    issue(0, 0);
    for (int k = 0; k < KSTEPS; ++k) {
        const int cur = k & 1;
        asm volatile("s_wait_asynccnt 0" ::: "memory");
        __syncthreads();
        if (k + 1 < KSTEPS) issue(cur ^ 1, k + 1);

        const char* base = smem + cur * BUF_B;
        const int   sub  = (l >> 4) * 16;   // K-half byte offset
        v16bf afr[4], bfr[2];
#pragma unroll
        for (int fm = 0; fm < 4; ++fm)
            afr[fm] = ldfrag(base + (wm + fm * 16 + (l & 15)) * ROWB + sub);
#pragma unroll
        for (int fn = 0; fn < 2; ++fn)
            bfr[fn] = ldfrag(base + ATILE_B + (wn + fn * 16 + (l & 15)) * ROWB + sub);

#pragma unroll
        for (int fm = 0; fm < 4; ++fm)
#pragma unroll
            for (int fn = 0; fn < 2; ++fn)
                acc[fm][fn] = __builtin_amdgcn_wmma_f32_16x16x32_bf16(
                    false, afr[fm], false, bfr[fn], (short)0, acc[fm][fn],
                    false, false);
        __syncthreads();
    }

    // epilogue: D layout -> VGPR j holds M = j + 8*(l>>4), N = l&15
    const int nl = l & 15;
    const int mh = (l >> 4) * 8;
#pragma unroll
    for (int fn = 0; fn < 2; ++fn) {
        const int col = o0 + wn + fn * 16 + nl;
        const float bv = bias[col];
#pragma unroll
        for (int fm = 0; fm < 4; ++fm)
#pragma unroll
            for (int j = 0; j < 8; ++j)
                out[(size_t)(b0 + wm + fm * 16 + mh + j) * OUT_ + col] =
                    acc[fm][fn][j] + bv;
    }
}

// ================= fallback: LDS-gather path (no workspace needed) =================
#define ROWS     16
#define STRIDE   4100
#define OCHUNK   1024
#define NTHREADS 256

__global__ __launch_bounds__(NTHREADS)
void condensed_gather_kernel(const float* __restrict__ in,
                             const float* __restrict__ w,
                             const float* __restrict__ bias,
                             const long long* __restrict__ mask64,
                             float* __restrict__ out)
{
    extern __shared__ float fsmem[];
    float* tile = fsmem;
    float* sbuf = fsmem + ROWS * STRIDE;

    const int tid = threadIdx.x;
    const int b0  = blockIdx.x * ROWS;
    const int o0  = blockIdx.y * OCHUNK;

    const float* src = in + (size_t)b0 * IN_;
    for (int q = tid; q < ROWS * (IN_ / 4); q += NTHREADS) {
        const int r = q >> 10;
        const int c = (q & 1023) << 2;
        const float* g = src + r * IN_ + c;
        unsigned lds = (unsigned)(uintptr_t)(tile + r * STRIDE + c);
        asm volatile("global_load_async_to_lds_b128 %0, %1, off"
                     :: "v"(lds), "v"(g) : "memory");
    }
    asm volatile("s_wait_asynccnt 0" ::: "memory");
    __syncthreads();

    const int b  = tid & 15;
    const int og = tid >> 4;
    const float* tb = tile + b * STRIDE;

    for (int j = 0; j < OCHUNK / 16; ++j) {
        const int o = o0 + j * 16 + og;
        const longlong2* ip = (const longlong2*)(mask64 + (size_t)o * FAN_);
        const float4*    wp = (const float4*)   (w      + (size_t)o * FAN_);
        float acc = 0.f;
#pragma unroll
        for (int q = 0; q < FAN_ / 4; ++q) {
            const longlong2 i0 = ip[2 * q];
            const longlong2 i1 = ip[2 * q + 1];
            const float4    wvv = wp[q];
            acc = fmaf(tb[(int)i0.x], wvv.x, acc);
            acc = fmaf(tb[(int)i0.y], wvv.y, acc);
            acc = fmaf(tb[(int)i1.x], wvv.z, acc);
            acc = fmaf(tb[(int)i1.y], wvv.w, acc);
        }
        acc += bias[o];
        sbuf[og * 17 + b] = acc;
        __syncthreads();
        out[(size_t)(b0 + (tid >> 4)) * OUT_ + (o0 + j * 16 + (tid & 15))] =
            sbuf[(tid & 15) * 17 + (tid >> 4)];
        __syncthreads();
    }
}

// ================= host =================
extern "C" void kernel_launch(void* const* d_in, const int* in_sizes, int n_in,
                              void* d_out, int out_size, void* d_ws, size_t ws_size,
                              hipStream_t stream)
{
    const float*     in   = (const float*)d_in[0];
    const float*     w    = (const float*)d_in[1];
    const float*     bias = (const float*)d_in[2];
    const long long* mask = (const long long*)d_in[3];
    float* out = (float*)d_out;

    const size_t st_bytes = (size_t)OUT_ * IN_ * sizeof(__bf16);   // 32 MB
    const size_t x_bytes  = (size_t)B_   * IN_ * sizeof(__bf16);   // 16 MB

    if (ws_size >= st_bytes + x_bytes) {
        __bf16* ST  = (__bf16*)d_ws;
        __bf16* Xbf = (__bf16*)((char*)d_ws + st_bytes);

        hipMemsetAsync(ST, 0, st_bytes, stream);

        const int n4 = (B_ * IN_) / 4;
        cvt_bf16_kernel<<<(n4 + 255) / 256, 256, 0, stream>>>(in, Xbf, n4);

        scatter_kernel<<<(OUT_ + 255) / 256, 256, 0, stream>>>(mask, w, ST);

        dim3 grid(OUT_ / TILE_N, B_ / TILE_M);   // (32, 16)
        wmma_gemm_kernel<<<grid, 256, GEMM_LDS, stream>>>(Xbf, ST, bias, out);
    } else {
        const size_t smem = (size_t)(ROWS * STRIDE + 16 * 17) * sizeof(float);
        dim3 grid(B_ / ROWS, OUT_ / OCHUNK);
        condensed_gather_kernel<<<grid, NTHREADS, smem, stream>>>(in, w, bias, mask, out);
    }
}
